// IDCF_LGCN_57887569215726
// MI455X (gfx1250) — compile-verified, matrix operand order checked by reading, not run on hardware
//
#include <hip/hip_runtime.h>
#include <hip/hip_bf16.h>

typedef __attribute__((ext_vector_type(16))) _Float16 v16h;
typedef __attribute__((ext_vector_type(8)))  float    v8f;

#define N_TOT 150000
#define N_USR 100000
#define DIM   64
#define NH    4
#define NS    50

// ---------------- WMMA fragment helpers (CDNA5 16-bit layouts, wave32) ----
// A (16x32 f16): lane%16 = M, lane/16 = K-half; VGPR v holds K pair
//   k(v) = 2*(v&3) + (v>=4?16:0) + 8*(lane/16)   (+ kc*32 chunk)
// B (32x16 f16): identical addressing with lane%16 = N, source stored so
//   that K is the contiguous (minor) axis.
__device__ inline v16h load_frag_f32(const float* p0, int ld, int kc, int lane) {
  int r = lane & 15, kh = lane >> 4;
  const float* p = p0 + (size_t)r * ld + kc * 32 + kh * 8;
  v16h f;
#pragma unroll
  for (int v = 0; v < 8; ++v) {
    int k = ((v & 3) << 1) + ((v & 4) ? 16 : 0);
    f[2 * v]     = (_Float16)p[k];
    f[2 * v + 1] = (_Float16)p[k + 1];
  }
  return f;
}

__device__ inline v16h load_frag_f16(const _Float16* p0, int ld, int kc, int lane) {
  int r = lane & 15, kh = lane >> 4;
  const _Float16* p = p0 + (size_t)r * ld + kc * 32 + kh * 8;
  v16h f;
#pragma unroll
  for (int v = 0; v < 8; ++v) {
    int k = ((v & 3) << 1) + ((v & 4) ? 16 : 0);
    f[2 * v]     = p[k];
    f[2 * v + 1] = p[k + 1];
  }
  return f;
}

__device__ inline v8f wmma16(v16h a, v16h b, v8f c) {
  return __builtin_amdgcn_wmma_f32_16x16x32_f16(
      /*neg_a=*/false, a, /*neg_b=*/false, b,
      /*c_mod=*/(short)0, c, /*reuse_a=*/false, /*reuse_b=*/false);
}

// Async global->LDS copy of 8 KB (wave32: 32 lanes x 16 B x 16 issues).
// Tracked by ASYNCcnt; caller must s_wait_asynccnt before reading the LDS.
__device__ inline void async_copy_8k(const _Float16* gsrc, _Float16* lds_dst, int lane) {
  unsigned lbase = (unsigned)(size_t)lds_dst + (unsigned)(lane * 16);
  unsigned long long gbase =
      (unsigned long long)(size_t)gsrc + (unsigned long long)(lane * 16);
#pragma unroll
  for (int it = 0; it < 16; ++it) {
    unsigned la = lbase + (unsigned)(it * 512);
    unsigned long long ga = gbase + (unsigned long long)(it * 512);
    asm volatile("global_load_async_to_lds_b128 %0, %1, off"
                 :: "v"(la), "v"(ga) : "memory");
  }
}

__device__ inline void wait_async0() {
  asm volatile("s_wait_asynccnt 0x0" ::: "memory");
}

// ---------------- SpMM scatter: y[row] += val * x[col], D=64 --------------
__global__ void spmm_kernel(const int* __restrict__ row, const int* __restrict__ col,
                            const float* __restrict__ val, const float* __restrict__ x,
                            float* __restrict__ y, int E) {
  long long gid = (long long)blockIdx.x * blockDim.x + threadIdx.x;
  long long e = gid >> 4;
  int part = (int)(gid & 15);
  if (e >= E) return;
  int r = row[e], c = col[e];
  float v = val[e];
  const float* xs = x + (size_t)c * DIM + part * 4;
  float* yd = y + (size_t)r * DIM + part * 4;
#pragma unroll
  for (int j = 0; j < 4; ++j) atomicAdd(yd + j, v * xs[j]);
}

// ---------------- precompute: K/V per (head,side) + f16 weight copies -----
// K16[(sh*64+n)*64+k] = su[n] . Wk_h[k] + bk_h[k]  (n>=NS -> 0)  [k-contig]
// V16[(sh*64+n)*64+k] = su[k] . Wv_h[n]            (k>=NS -> 0)  [k-contig,
//                       i.e. stored transposed so B loads are contiguous]
__global__ void precompute_kernel(const float* __restrict__ emb,
                                  const int* __restrict__ susr,
                                  const int* __restrict__ sitm,
                                  const float* __restrict__ wq,
                                  const float* __restrict__ wk,
                                  const float* __restrict__ bk,
                                  const float* __restrict__ wv,
                                  const float* __restrict__ wo,
                                  _Float16* __restrict__ wq16,
                                  _Float16* __restrict__ wo16,
                                  _Float16* __restrict__ K16,
                                  _Float16* __restrict__ V16) {
  int b = blockIdx.x, tid = threadIdx.x;
  if (b == 8) {
    for (int i = tid; i < NH * 64 * 64; i += blockDim.x) wq16[i] = (_Float16)wq[i];
    return;
  }
  if (b == 9) {
    for (int i = tid; i < 64 * 256; i += blockDim.x) wo16[i] = (_Float16)wo[i];
    return;
  }
  int side = b & 1, h = b >> 1;
  const int* idx = side ? sitm : susr;
  int off = side ? N_USR : 0;
  int sh = side * NH + h;
  for (int t = tid; t < 64 * 64; t += blockDim.x) {
    int i = t >> 6, j = t & 63;
    float kv = 0.f;
    if (i < NS) {
      int r = off + idx[h * NS + i];
      const float* e = emb + (size_t)r * DIM;
      const float* w = wk + (size_t)(h * 64 + j) * DIM;
      float d = 0.f;
      for (int c = 0; c < DIM; ++c) d += e[c] * w[c];
      kv = d + bk[h * 64 + j];
    }
    K16[((size_t)sh * 64 + i) * 64 + j] = (_Float16)kv;
    float vv = 0.f;
    if (j < NS) {
      int r = off + idx[h * NS + j];
      const float* e = emb + (size_t)r * DIM;
      const float* w = wv + (size_t)(h * 64 + i) * DIM;
      float d = 0.f;
      for (int c = 0; c < DIM; ++c) d += e[c] * w[c];
      vv = d;
    }
    V16[((size_t)sh * 64 + i) * 64 + j] = (_Float16)vv;
  }
}

// ---------------- fused GAT (4 heads) + output projection -----------------
// One wave per 16-row tile. Per head: async-stage K/V slices into LDS while
// the q = x@Wq^T WMMAs run, then softmax(q k^T) V' @ Wout_h^T accumulation.
__global__ void __launch_bounds__(32)
gat_kernel(const float* __restrict__ xq,
           const _Float16* __restrict__ wq16,
           const _Float16* __restrict__ wo16,
           const _Float16* __restrict__ K16,
           const _Float16* __restrict__ V16,
           const float* __restrict__ bq,
           const float* __restrict__ bv,
           const float* __restrict__ bo,
           float* __restrict__ ra,
           float* __restrict__ out) {
  __shared__ _Float16 kbuf[64 * 64];   // async-staged K slice (8 KB)
  __shared__ _Float16 vbuf[64 * 64];   // async-staged V slice (8 KB)
  __shared__ _Float16 q16[16 * 64];
  __shared__ _Float16 a16[16 * 64];
  __shared__ _Float16 o16[16 * 64];
  __shared__ float    s32[16 * 64];

  const int lane = threadIdx.x;
  const int base = blockIdx.x * 16;
  const int side = (base >= N_USR) ? 1 : 0;
  const int n = lane & 15, kh = lane >> 4;

  v16h ax[2];
#pragma unroll
  for (int kc = 0; kc < 2; ++kc)
    ax[kc] = load_frag_f32(xq + (size_t)base * DIM, DIM, kc, lane);

  v8f crep[4];
#pragma unroll
  for (int nt = 0; nt < 4; ++nt) { v8f z = {}; crep[nt] = z; }

  for (int h = 0; h < NH; ++h) {
    const int sh = side * NH + h;
    // kick off async K/V staging; overlaps with the q-projection WMMAs
    async_copy_8k(K16 + (size_t)sh * 64 * 64, kbuf, lane);
    async_copy_8k(V16 + (size_t)sh * 64 * 64, vbuf, lane);

    // q = x @ Wq_h^T + bq_h
#pragma unroll
    for (int nt = 0; nt < 4; ++nt) {
      v8f acc = {};
#pragma unroll
      for (int kc = 0; kc < 2; ++kc)
        acc = wmma16(ax[kc], load_frag_f16(wq16 + (size_t)(h * 64 + nt * 16) * 64, 64, kc, lane), acc);
      int col = nt * 16 + n;
      float bqv = bq[h * 64 + col];
#pragma unroll
      for (int v = 0; v < 8; ++v)
        q16[(v + 8 * kh) * 64 + col] = (_Float16)(acc[v] + bqv);
    }
    __syncthreads();
    v16h aq[2];
#pragma unroll
    for (int kc = 0; kc < 2; ++kc) aq[kc] = load_frag_f16(q16, 64, kc, lane);

    wait_async0();  // K/V now resident in LDS
    // scores = q @ K_h^T  (cols >= NS are padding, masked in softmax)
#pragma unroll
    for (int nt = 0; nt < 4; ++nt) {
      v8f acc = {};
#pragma unroll
      for (int kc = 0; kc < 2; ++kc)
        acc = wmma16(aq[kc], load_frag_f16(kbuf + (size_t)(nt * 16) * 64, 64, kc, lane), acc);
      int col = nt * 16 + n;
#pragma unroll
      for (int v = 0; v < 8; ++v)
        s32[(v + 8 * kh) * 64 + col] = acc[v];
    }
    __syncthreads();
    // row softmax over the NS valid columns
    if (lane < 16) {
      float mx = -1e30f;
      for (int c = 0; c < NS; ++c) mx = fmaxf(mx, s32[lane * 64 + c]);
      float sum = 0.f;
      for (int c = 0; c < NS; ++c) {
        float e = __expf(s32[lane * 64 + c] - mx);
        s32[lane * 64 + c] = e;
        sum += e;
      }
      float inv = 1.f / sum;
      for (int c = 0; c < NS; ++c) a16[lane * 64 + c] = (_Float16)(s32[lane * 64 + c] * inv);
      for (int c = NS; c < 64; ++c) a16[lane * 64 + c] = (_Float16)0.f;
    }
    __syncthreads();
    v16h aa[2];
#pragma unroll
    for (int kc = 0; kc < 2; ++kc) aa[kc] = load_frag_f16(a16, 64, kc, lane);
    // out_h = attn @ V'_h + bv_h
#pragma unroll
    for (int nt = 0; nt < 4; ++nt) {
      v8f acc = {};
#pragma unroll
      for (int kc = 0; kc < 2; ++kc)
        acc = wmma16(aa[kc], load_frag_f16(vbuf + (size_t)(nt * 16) * 64, 64, kc, lane), acc);
      int col = nt * 16 + n;
      float bvv = bv[h * 64 + col];
#pragma unroll
      for (int v = 0; v < 8; ++v)
        o16[(v + 8 * kh) * 64 + col] = (_Float16)(acc[v] + bvv);
    }
    __syncthreads();
    v16h ao[2];
#pragma unroll
    for (int kc = 0; kc < 2; ++kc) ao[kc] = load_frag_f16(o16, 64, kc, lane);
    // rep += out_h @ Wout[:, h*64:(h+1)*64]^T
#pragma unroll
    for (int nt = 0; nt < 4; ++nt) {
#pragma unroll
      for (int kc = 0; kc < 2; ++kc)
        crep[nt] = wmma16(ao[kc],
                          load_frag_f16(wo16 + (size_t)(nt * 16) * 256 + h * 64, 256, kc, lane),
                          crep[nt]);
    }
    __syncthreads();
  }
  // rep += bo; store to propagation buffer and into the accumulator (d_out)
#pragma unroll
  for (int nt = 0; nt < 4; ++nt) {
    int col = nt * 16 + n;
    float bov = bo[col];
#pragma unroll
    for (int v = 0; v < 8; ++v) {
      int row = base + v + 8 * kh;
      float val = crep[nt][v] + bov;
      ra[(size_t)row * DIM + col] = val;
      out[(size_t)row * DIM + col] = val;
    }
  }
}

__global__ void add_kernel(float* __restrict__ out, const float* __restrict__ src, long long nelem) {
  long long i = (long long)blockIdx.x * blockDim.x + threadIdx.x;
  if (i < nelem) out[i] += src[i];
}

__global__ void scale_kernel(float* __restrict__ out, float s, long long nelem) {
  long long i = (long long)blockIdx.x * blockDim.x + threadIdx.x;
  if (i < nelem) out[i] *= s;
}

extern "C" void kernel_launch(void* const* d_in, const int* in_sizes, int n_in,
                              void* d_out, int out_size, void* d_ws, size_t ws_size,
                              hipStream_t stream) {
  (void)n_in; (void)out_size; (void)ws_size;
  const float* emb  = (const float*)d_in[0];
  const int*   arow = (const int*)d_in[1];
  const int*   acol = (const int*)d_in[2];
  const float* aval = (const float*)d_in[3];
  const int*   frow = (const int*)d_in[4];
  const int*   fcol = (const int*)d_in[5];
  const float* fval = (const float*)d_in[6];
  const int*   susr = (const int*)d_in[7];
  const int*   sitm = (const int*)d_in[8];
  const float* wq   = (const float*)d_in[9];
  const float* bq   = (const float*)d_in[10];
  const float* wk   = (const float*)d_in[11];
  const float* bk   = (const float*)d_in[12];
  const float* wv   = (const float*)d_in[13];
  const float* bv   = (const float*)d_in[14];
  const float* wo   = (const float*)d_in[15];
  const float* bo   = (const float*)d_in[16];
  float* out = (float*)d_out;

  const long long nd = (long long)N_TOT * DIM;
  float* xq = (float*)d_ws;              // 38.4 MB: x_q, later reused as r ping-pong
  float* ra = xq + nd;                   // 38.4 MB: rep / r ping-pong
  _Float16* wq16 = (_Float16*)(ra + nd);
  _Float16* wo16 = wq16 + NH * 64 * 64;
  _Float16* K16  = wo16 + 64 * 256;
  _Float16* V16  = K16 + 2 * NH * 64 * 64;
  float* rb = xq;                        // alias: x_q dead after gat_kernel

  int Ea = in_sizes[1];
  int Ef = in_sizes[4];

  // x_q = A_feat @ embedding
  hipMemsetAsync(xq, 0, nd * sizeof(float), stream);
  {
    long long th = (long long)Ef * 16;
    spmm_kernel<<<(int)((th + 255) / 256), 256, 0, stream>>>(frow, fcol, fval, emb, xq, Ef);
  }
  // per-head K/V matrices + f16 weights
  precompute_kernel<<<10, 256, 0, stream>>>(emb, susr, sitm, wq, wk, bk, wv, wo,
                                            wq16, wo16, K16, V16);
  // fused GAT + output projection -> ra (= rep) and d_out (= acc)
  gat_kernel<<<N_TOT / 16, 32, 0, stream>>>(xq, wq16, wo16, K16, V16, bq, bv, bo, ra, out);
  // 3x propagation: r = A_adj @ r ; acc += r
  float* src = ra;
  float* dst = rb;
  for (int l = 0; l < 3; ++l) {
    hipMemsetAsync(dst, 0, nd * sizeof(float), stream);
    long long th = (long long)Ea * 16;
    spmm_kernel<<<(int)((th + 255) / 256), 256, 0, stream>>>(arow, acol, aval, src, dst, Ea);
    add_kernel<<<(int)((nd + 255) / 256), 256, 0, stream>>>(out, dst, nd);
    float* t = src; src = dst; dst = t;
  }
  scale_kernel<<<(int)((nd + 255) / 256), 256, 0, stream>>>(out, 0.25f, nd);
}
